// ExampleModel_9706626088960
// MI455X (gfx1250) — matrix operand order repaired and checked by reading, not compiled, not run on hardware
//
#include <hip/hip_runtime.h>
#include <hip/hip_bf16.h>
#include <math.h>
#include <stdint.h>

// ---------------------------------------------------------------------------
// MoE (top-2, E=8) forward collapsed to per-token scalar + log_softmax.
// B=4 S=512 D=1024 H=2048 E=8 k=2, T=B*S=2048.
//   s[t] = sum_{e in top2(t)} g[t,e] * ( relu(x[t]@w1[e]+b1[e]) . w2s[e] + b2s[e] )
//   out[b,s] = log_softmax_S( s )
// w2s[e][h] = sum_d w2[e][h][d],  b2s[e] = sum_d b2[e][d]  (precomputed).
// Heavy GEMM in bf16 WMMA (v_wmma_f32_16x16x32_bf16), fp32 accumulate.
// Token-row gather into LDS done by the Tensor Data Mover in gather mode.
// ---------------------------------------------------------------------------

typedef __bf16 bf16;
typedef __attribute__((ext_vector_type(16))) __bf16 v16bf;
typedef __attribute__((ext_vector_type(8)))  float  v8f;
typedef __attribute__((ext_vector_type(4)))  unsigned int v4u;
typedef __attribute__((ext_vector_type(8)))  int    v8i;
typedef __attribute__((ext_vector_type(4)))  int    v4i;

#define T_TOK 2048
#define DIM   1024
#define HID   2048
#define NEXP  8
#define BATCH 4
#define SEQ   512

#define HB          128          // h-columns per FFN block
#define LDS_STRIDE  1040         // 1024 + 16 pad elems (matches TDM pad 4DW/256DW)

// workspace layout (bytes)
#define OFF_W2S 0                                   // E*H fp32
#define OFF_B2S (OFF_W2S + NEXP*HID*4)              // E fp32 (+pad)
#define OFF_CNT (OFF_B2S + 64)                      // E int  (+pad)
#define OFF_S   (OFF_CNT + 64)                      // T fp32
#define OFF_IDX (OFF_S   + T_TOK*4)                 // E*T int
#define OFF_VAL (OFF_IDX + NEXP*T_TOK*4)            // E*T fp32
#define OFF_XB  (OFF_VAL + NEXP*T_TOK*4)            // T*D bf16
#define OFF_W1B (OFF_XB  + T_TOK*DIM*2)             // E*D*H bf16

#define FFN_SMEM (HB*LDS_STRIDE*2 + 16*LDS_STRIDE*2 + 8*16*4 + 16*4 + 16*4 + 64)

// ---------------------------------------------------------------- utilities
__global__ void zero_cnt_kernel(int* __restrict__ cnt) {
    if (threadIdx.x < NEXP) cnt[threadIdx.x] = 0;
}

// one wave per row: rows [0, E*H) reduce w2 rows, rows [E*H, E*H+E) reduce b2
__global__ void prep_kernel(const float* __restrict__ w2, const float* __restrict__ b2,
                            float* __restrict__ w2s, float* __restrict__ b2s) {
    const int wave = threadIdx.x >> 5, lane = threadIdx.x & 31;
    const int row  = blockIdx.x * (blockDim.x >> 5) + wave;
    const int NR   = NEXP * HID;
    if (row >= NR + NEXP) return;
    const float* src = (row < NR) ? (w2 + (size_t)row * DIM)
                                  : (b2 + (size_t)(row - NR) * DIM);
    float v = 0.f;
    for (int d = lane; d < DIM; d += 32) v += src[d];
    v += __shfl_xor(v, 16); v += __shfl_xor(v, 8);
    v += __shfl_xor(v, 4);  v += __shfl_xor(v, 2); v += __shfl_xor(v, 1);
    if (lane == 0) { if (row < NR) w2s[row] = v; else b2s[row - NR] = v; }
}

__global__ void cvt_kernel(const float* __restrict__ src, bf16* __restrict__ dst, int n) {
    int i = blockIdx.x * blockDim.x + threadIdx.x;
    const int stride = gridDim.x * blockDim.x;
    for (; i < n; i += stride) dst[i] = (bf16)src[i];
}

// ------------------------------------------------------------------- gate
// one wave per token: fp32 logits, softmax, top-2, build per-expert lists,
// and seed s[t] with the (gate-weighted) b2 column-sum contribution.
__global__ void gate_kernel(const float* __restrict__ x, const float* __restrict__ wg,
                            const float* __restrict__ b2s, int* __restrict__ cnt,
                            int* __restrict__ idx, float* __restrict__ val,
                            float* __restrict__ s) {
    const int wave = threadIdx.x >> 5, lane = threadIdx.x & 31;
    const int t = blockIdx.x * (blockDim.x >> 5) + wave;
    if (t >= T_TOK) return;

    float acc[NEXP];
#pragma unroll
    for (int e = 0; e < NEXP; ++e) acc[e] = 0.f;

    const float* xr = x + (size_t)t * DIM;
    for (int d = lane; d < DIM; d += 32) {
        const float xv = xr[d];
        const float* wr = wg + d * NEXP;
#pragma unroll
        for (int e = 0; e < NEXP; ++e) acc[e] += xv * wr[e];
    }
#pragma unroll
    for (int e = 0; e < NEXP; ++e) {
        float v = acc[e];
        v += __shfl_xor(v, 16); v += __shfl_xor(v, 8);
        v += __shfl_xor(v, 4);  v += __shfl_xor(v, 2); v += __shfl_xor(v, 1);
        acc[e] = v;
    }
    if (lane == 0) {
        float m = acc[0];
#pragma unroll
        for (int e = 1; e < NEXP; ++e) m = fmaxf(m, acc[e]);
        float p[NEXP], sum = 0.f;
#pragma unroll
        for (int e = 0; e < NEXP; ++e) { p[e] = __expf(acc[e] - m); sum += p[e]; }
        const float inv = 1.f / sum;
#pragma unroll
        for (int e = 0; e < NEXP; ++e) p[e] *= inv;
        // top-2 (k==2 in reference)
        int i1 = 0;
#pragma unroll
        for (int e = 1; e < NEXP; ++e) if (p[e] > p[i1]) i1 = e;
        int i2 = (i1 == 0) ? 1 : 0;
#pragma unroll
        for (int e = 0; e < NEXP; ++e) if (e != i1 && p[e] > p[i2]) i2 = e;

        int s1 = atomicAdd(&cnt[i1], 1);
        idx[i1 * T_TOK + s1] = t;  val[i1 * T_TOK + s1] = p[i1];
        int s2 = atomicAdd(&cnt[i2], 1);
        idx[i2 * T_TOK + s2] = t;  val[i2 * T_TOK + s2] = p[i2];
        s[t] = p[i1] * b2s[i1] + p[i2] * b2s[i2];
    }
}

// -------------------------------------------------------------------- FFN
// grid = (H/HB, E). Block stages w1[e][:, hb0:hb0+HB) transposed in LDS,
// loops over this expert's 16-token tiles; wave w owns h-tile w (16 cols).
// Token rows of x are gathered into LDS by the TDM (gather mode, 16-bit idx),
// with pad_interval=256DW/pad_amount=4DW so rows land at LDS_STRIDE=1040.
__global__ void ffn_kernel(const bf16* __restrict__ xb, const bf16* __restrict__ w1b,
                           const float* __restrict__ b1, const float* __restrict__ w2s,
                           const int* __restrict__ cnt, const int* __restrict__ idx,
                           const float* __restrict__ val, float* __restrict__ s) {
    extern __shared__ char smem[];
    bf16*  w1s   = (bf16*)smem;                                   // [HB][LDS_STRIDE]
    bf16*  xs    = (bf16*)(smem + HB * LDS_STRIDE * 2);           // [16][LDS_STRIDE]
    float* red   = (float*)(smem + (HB + 16) * LDS_STRIDE * 2);   // [8][16]
    int*   toks  = (int*)(red + 8 * 16);                          // [16]
    float* gvals = (float*)(toks + 16);                           // [16]
    __shared__ int ntok;

    const int e   = blockIdx.y;
    const int hb0 = blockIdx.x * HB;
    const int tid = threadIdx.x;

    // stage w1 slice transposed: w1s[col][d]  (contiguous-k B fragments)
    for (int i = tid; i < DIM * HB; i += blockDim.x) {
        const int d = i >> 7;           // / HB
        const int c = i & (HB - 1);
        w1s[c * LDS_STRIDE + d] = w1b[((size_t)e * DIM + d) * HID + hb0 + c];
    }
    if (tid == 0) ntok = cnt[e];
    __syncthreads();

    const int n_t    = ntok;
    const int ntiles = (n_t + 15) >> 4;
    const int wave = tid >> 5, lane = tid & 31;
    const int half = lane >> 4, nn = lane & 15;
    const int hl   = wave * 16 + nn;                 // local h column (0..127)
    const float b1v  = b1[e * HID + hb0 + hl];
    const float w2sv = w2s[e * HID + hb0 + hl];

    // ---- TDM gather descriptor, loop-invariant groups 0/1 (ISA 8.3/8.4/8.7)
    const unsigned long long ga = (unsigned long long)(uintptr_t)xb;
    v4u g0;
    g0.x = 0x80000001u;                          // gather_mode=1, 16b idx, count=1
    g0.y = (unsigned)(uintptr_t)xs;              // lds_addr (bytes)
    g0.z = (unsigned)(ga & 0xFFFFFFFFull);       // global_addr[31:0]
    g0.w = (unsigned)((ga >> 32) & 0x01FFFFFFull) | 0x80000000u;   // [56:32] | type=2
    v8i g1;
    g1[0] = 0x07D10000;   // data_size=2B, pad_enable, pad_interval=7(256DW), pad_amt=3(4DW)
    g1[1] = 0x04000000;   // tensor_dim0[15:0]=1024 in bits 63:48
    g1[2] = 0x08000000;   // tensor_dim1[15:0]=2048 in bits 95:80
    g1[3] = 0x04000000;   // tile_dim0=1024 in bits 127:112
    g1[4] = 0x00000010;   // tile_dim1 = 16 gather indices
    g1[5] = 0x00000400;   // tensor_dim0_stride = 1024 (low 32)
    g1[6] = 0;
    g1[7] = 0;
    const v8i gz = {0, 0, 0, 0, 0, 0, 0, 0};     // trailing zero group (6-arg form)

    for (int tile = 0; tile < ntiles; ++tile) {
        if (tid < 16) {
            const int slot = tile * 16 + tid;
            int tok = 0; float gv = 0.f;
            if (slot < n_t) { tok = idx[e * T_TOK + slot]; gv = val[e * T_TOK + slot]; }
            toks[tid] = tok; gvals[tid] = gv;
        }
        __syncthreads();
        // wave 0 issues the TDM gather of the 16 token rows into xs
        if (wave == 0) {
            v4i g2, g3;
#pragma unroll
            for (int j = 0; j < 4; ++j)
                g2[j] = (toks[2 * j] & 0xFFFF) | (toks[2 * j + 1] << 16);
#pragma unroll
            for (int j = 0; j < 4; ++j)
                g3[j] = (toks[8 + 2 * j] & 0xFFFF) | (toks[8 + 2 * j + 1] << 16);
            __builtin_amdgcn_tensor_load_to_lds(g0, g1, g2, g3, gz, 0);
            __builtin_amdgcn_s_wait_tensorcnt(0);
        }
        __syncthreads();

        v8f c = {};
        const bf16* xrow = &xs[nn * LDS_STRIDE];
#pragma unroll 4
        for (int kk = 0; kk < DIM; kk += 32) {
            // TDM pad inserts 8 elems after the first 512 of each row
            const bf16* xr = xrow + kk + ((kk >= 512) ? 8 : 0);
            // A fragment: contiguous-k (lanes<16: K 0-15; lanes>=16: K 16-31)
            v16bf a;
#pragma unroll
            for (int p = 0; p < 8; ++p) {
                const int kb = (p << 1) + half * 16;
                a[2 * p]     = xr[kb];
                a[2 * p + 1] = xr[kb + 1];
            }
            // B fragment: contiguous-k, same order as A (lanes<16: K 0-15; lanes>=16: K 16-31)
            v16bf b;
            const bf16* wr = &w1s[hl * LDS_STRIDE + kk + half * 16];
#pragma unroll
            for (int q = 0; q < 16; ++q) b[q] = wr[q];

            c = __builtin_amdgcn_wmma_f32_16x16x32_bf16(
                    false, a, false, b, (short)0, c, false, false);
        }
        // epilogue: bias + relu + w2s scale, reduce over the 16 columns
#pragma unroll
        for (int r = 0; r < 8; ++r) {
            float hv = c[r] + b1v;
            hv = hv > 0.f ? hv : 0.f;
            float v = hv * w2sv;
            v += __shfl_xor(v, 1); v += __shfl_xor(v, 2);
            v += __shfl_xor(v, 4); v += __shfl_xor(v, 8);
            if (nn == 0) red[wave * 16 + half * 8 + r] = v;   // token m = half*8+r
        }
        __syncthreads();
        if (tid < 16) {
            float tot = 0.f;
#pragma unroll
            for (int w = 0; w < 8; ++w) tot += red[w * 16 + tid];
            const float gv = gvals[tid];
            if (gv != 0.f) atomicAdd(&s[toks[tid]], tot * gv);
        }
        __syncthreads();
    }
}

// ------------------------------------------------------------ log_softmax
__global__ void lsm_kernel(const float* __restrict__ s, float* __restrict__ out) {
    __shared__ float buf[SEQ];
    const int b = blockIdx.x, tid = threadIdx.x;
    const float v = s[b * SEQ + tid];
    buf[tid] = v; __syncthreads();
    for (int off = SEQ / 2; off > 0; off >>= 1) {
        if (tid < off) buf[tid] = fmaxf(buf[tid], buf[tid + off]);
        __syncthreads();
    }
    const float m = buf[0]; __syncthreads();
    buf[tid] = __expf(v - m); __syncthreads();
    for (int off = SEQ / 2; off > 0; off >>= 1) {
        if (tid < off) buf[tid] += buf[tid + off];
        __syncthreads();
    }
    out[b * SEQ + tid] = v - m - __logf(buf[0]);
}

// ------------------------------------------------------------------- host
extern "C" void kernel_launch(void* const* d_in, const int* in_sizes, int n_in,
                              void* d_out, int out_size, void* d_ws, size_t ws_size,
                              hipStream_t stream) {
    (void)in_sizes; (void)n_in; (void)out_size; (void)ws_size;
    const float* x  = (const float*)d_in[0];
    const float* wg = (const float*)d_in[1];
    const float* w1 = (const float*)d_in[2];
    const float* b1 = (const float*)d_in[3];
    const float* w2 = (const float*)d_in[4];
    const float* b2 = (const float*)d_in[5];

    char*  ws  = (char*)d_ws;
    float* w2s = (float*)(ws + OFF_W2S);
    float* b2s = (float*)(ws + OFF_B2S);
    int*   cnt = (int*)  (ws + OFF_CNT);
    float* s   = (float*)(ws + OFF_S);
    int*   idx = (int*)  (ws + OFF_IDX);
    float* val = (float*)(ws + OFF_VAL);
    bf16*  xb  = (bf16*) (ws + OFF_XB);
    bf16*  w1b = (bf16*) (ws + OFF_W1B);

    zero_cnt_kernel<<<1, 32, 0, stream>>>(cnt);
    prep_kernel<<<(NEXP * HID + NEXP + 7) / 8, 256, 0, stream>>>(w2, b2, w2s, b2s);
    cvt_kernel<<<2048, 256, 0, stream>>>(x,  xb,  T_TOK * DIM);
    cvt_kernel<<<8192, 256, 0, stream>>>(w1, w1b, NEXP * DIM * HID);
    gate_kernel<<<T_TOK / 8, 256, 0, stream>>>(x, wg, b2s, cnt, idx, val, s);

    (void)hipFuncSetAttribute((const void*)ffn_kernel,
                              hipFuncAttributeMaxDynamicSharedMemorySize, FFN_SMEM);
    ffn_kernel<<<dim3(HID / HB, NEXP), 256, FFN_SMEM, stream>>>(
        xb, w1b, b1, w2s, cnt, idx, val, s);

    lsm_kernel<<<BATCH, SEQ, 0, stream>>>(s, (float*)d_out);
}
